// GraphAttentionNetwork_85564338471069
// MI455X (gfx1250) — matrix operand order repaired and checked by reading, not compiled, not run on hardware
//
#include <hip/hip_runtime.h>
#include <hip/hip_bf16.h>

// ---------- types ----------
typedef __attribute__((ext_vector_type(16))) __bf16          v16bf;
typedef __attribute__((ext_vector_type(8)))  float           v8f;
typedef __attribute__((ext_vector_type(8)))  unsigned short  u16x8;
typedef __attribute__((ext_vector_type(16))) unsigned short  u16x16;

__device__ __forceinline__ unsigned short f2bf(float f) {
    unsigned int u = __float_as_uint(f);
    unsigned int r = u + 0x7FFFu + ((u >> 16) & 1u);   // round-to-nearest-even
    return (unsigned short)(r >> 16);
}

// =====================================================================
// Kernel 1: per-block partial column sums of x [4096,128] -> part[32][128]
// =====================================================================
__global__ __launch_bounds__(256) void colmean_part(const float* __restrict__ x,
                                                    float* __restrict__ part) {
    __shared__ float s0[256];
    const int t   = threadIdx.x;
    const int b   = blockIdx.x;          // 32 blocks, 128 rows each
    const int col = t & 127;
    const int sub = t >> 7;              // 2 sub-groups of 64 rows
    const float* xp = x + (size_t)(b * 128 + sub * 64) * 128 + col;
    float s = 0.f;
    #pragma unroll 8
    for (int r = 0; r < 64; ++r) s += xp[(size_t)r * 128];
    s0[t] = s;
    __syncthreads();
    if (t < 128) part[b * 128 + t] = s0[t] + s0[t + 128];
}

// =====================================================================
// Kernel 2: reduce partials -> mean(x); u0 = elu(mean@W0^T+b0);
//           u1 = elu(u0@W1^T+b1); gi = u1@Wih^T+bih  (single block, 384 thr)
// =====================================================================
__global__ __launch_bounds__(384) void prep(const float* __restrict__ part,
                                            const float* __restrict__ W0,
                                            const float* __restrict__ b0,
                                            const float* __restrict__ W1,
                                            const float* __restrict__ b1,
                                            const float* __restrict__ Wih,
                                            const float* __restrict__ bih,
                                            float* __restrict__ gi_out) {
    __shared__ __align__(16) float mx[128];
    __shared__ __align__(16) float u0s[128];
    __shared__ __align__(16) float u1s[128];
    const int t = threadIdx.x;

    if (t < 128) {
        float s = 0.f;
        #pragma unroll
        for (int p = 0; p < 32; ++p) s += part[p * 128 + t];
        mx[t] = s * (1.0f / 4096.0f);
    }
    __syncthreads();

    if (t < 128) {                              // layer 0 matvec + ELU
        float s = b0[t];
        const float4* wr = (const float4*)(W0 + t * 128);
        const float4* xv = (const float4*)mx;
        #pragma unroll
        for (int k = 0; k < 32; ++k) {
            float4 w = wr[k]; float4 v = xv[k];
            s = fmaf(w.x, v.x, s); s = fmaf(w.y, v.y, s);
            s = fmaf(w.z, v.z, s); s = fmaf(w.w, v.w, s);
        }
        u0s[t] = (s > 0.f) ? s : (__expf(s) - 1.f);
    }
    __syncthreads();

    if (t < 128) {                              // layer 1 matvec + ELU
        float s = b1[t];
        const float4* wr = (const float4*)(W1 + t * 128);
        const float4* xv = (const float4*)u0s;
        #pragma unroll
        for (int k = 0; k < 32; ++k) {
            float4 w = wr[k]; float4 v = xv[k];
            s = fmaf(w.x, v.x, s); s = fmaf(w.y, v.y, s);
            s = fmaf(w.z, v.z, s); s = fmaf(w.w, v.w, s);
        }
        u1s[t] = (s > 0.f) ? s : (__expf(s) - 1.f);
    }
    __syncthreads();

    {                                           // gi = Wih @ u1 + bih  [384]
        float s = bih[t];
        const float4* wr = (const float4*)(Wih + t * 128);
        const float4* xv = (const float4*)u1s;
        #pragma unroll
        for (int k = 0; k < 32; ++k) {
            float4 w = wr[k]; float4 v = xv[k];
            s = fmaf(w.x, v.x, s); s = fmaf(w.y, v.y, s);
            s = fmaf(w.z, v.z, s); s = fmaf(w.w, v.w, s);
        }
        gi_out[t] = s;
    }
}

// =====================================================================
// Kernel 3: convert/pad Wout [66,128] -> bf16 [80,128] (rows 66..79 = 0)
// =====================================================================
__global__ __launch_bounds__(256) void wout_pad(const float* __restrict__ Wout,
                                                unsigned short* __restrict__ wb) {
    const int i = blockIdx.x * 256 + threadIdx.x;   // 80*128 = 10240
    if (i < 80 * 128) {
        const int row = i >> 7;
        const float v = (row < 66) ? Wout[i] : 0.f;
        wb[i] = f2bf(v);
    }
}

// =====================================================================
// Kernel 4: sequential GRU scan, 4096 steps, constant input gi.
// One block, 384 threads (12 waves on one WGP). Each thread's Whh row
// (128 f32) is hoisted into VGPRs ONCE; per-step traffic is only the
// h[128] vector via conflict-free LDS broadcast reads. Step cost ->
// VALU floor (~49K MACs / 128 lanes*SIMD = ~384 cyc). hs emitted as bf16.
// =====================================================================
__global__ __launch_bounds__(384) void gru_scan(const float* __restrict__ Whh,
                                                const float* __restrict__ bhh,
                                                const float* __restrict__ gi,
                                                unsigned short* __restrict__ hsb) {
    __shared__ __align__(16) float h[128];
    __shared__ __align__(16) float gh[384];
    const int t = threadIdx.x;

    const float bh = bhh[t];
    float gr = 0.f, gz = 0.f, gn = 0.f;
    if (t < 128) {
        gr = gi[t]; gz = gi[t + 128]; gn = gi[t + 256];
        h[t] = 0.f;
    }

    // Hoist this thread's Whh row into registers (loop-invariant weights).
    const float4* wrow = (const float4*)(Whh + t * 128);
    float4 w[32];
    #pragma unroll
    for (int k = 0; k < 32; ++k) w[k] = wrow[k];

    __syncthreads();

    for (int step = 0; step < 4096; ++step) {
        // gh[t] = dot(Whh[t,:], h) + bhh[t]
        float a0 = 0.f, a1 = 0.f, a2 = 0.f, a3 = 0.f;
        const float4* hv = (const float4*)h;
        #pragma unroll
        for (int k = 0; k < 32; ++k) {
            float4 v = hv[k];
            a0 = fmaf(w[k].x, v.x, a0); a1 = fmaf(w[k].y, v.y, a1);
            a2 = fmaf(w[k].z, v.z, a2); a3 = fmaf(w[k].w, v.w, a3);
        }
        gh[t] = (a0 + a1) + (a2 + a3) + bh;
        __syncthreads();

        if (t < 128) {                       // gates + state update
            const float hr = gh[t], hz = gh[t + 128], hn = gh[t + 256];
            const float r  = 1.f / (1.f + __expf(-(gr + hr)));
            const float z  = 1.f / (1.f + __expf(-(gz + hz)));
            const float nc = tanhf(gn + r * hn);
            const float hp = h[t];
            const float hnew = fmaf(z, hp - nc, nc);   // (1-z)*nc + z*hp
            h[t] = hnew;
            hsb[step * 128 + t] = f2bf(hnew);
        }
        __syncthreads();
    }
}

// =====================================================================
// Kernel 5: out = hs(bf16)[4096,128] @ Wout^T + bout via v_wmma_f32_16x16x32_bf16.
// One 16x16 output tile per wave32; 256 M-tiles x 5 N-tiles; 4 K-steps.
// A/B fragments use k = 16*(lane/16) + s in both (16-bit A 16x32; B 32x16; C/D 8 VGPR).
// =====================================================================
__global__ __launch_bounds__(128) void out_gemm(const unsigned short* __restrict__ hsb,
                                                const unsigned short* __restrict__ wb,
                                                const float* __restrict__ bout,
                                                float* __restrict__ out) {
    const int wave = threadIdx.x >> 5;
    const int lane = threadIdx.x & 31;
    const int tile = blockIdx.x * 4 + wave;        // 0..1279
    const int mt = tile / 5;
    const int nt = tile % 5;

    const int lo16 = lane & 15;
    const int hi   = lane >> 4;                    // half-wave select
    const int arow = mt * 16 + lo16;               // A: lane -> M
    const int wrow = nt * 16 + lo16;               // B: lane -> N (padded Wout row)

    v8f c = {};
    #pragma unroll
    for (int kb = 0; kb < 128; kb += 32) {
        // A (16x32 bf16): lo half K=kb+0..15, hi half K=kb+16..31
        u16x8 aLo = *(const u16x8*)(hsb + (size_t)arow * 128 + kb + hi * 16);
        u16x8 aHi = *(const u16x8*)(hsb + (size_t)arow * 128 + kb + hi * 16 + 8);
        u16x16 au;
        #pragma unroll
        for (int i = 0; i < 8; ++i) { au[i] = aLo[i]; au[8 + i] = aHi[i]; }
        // B (32x16 bf16): lane = N, lo half K=kb+0..15, hi half K=kb+16..31
        u16x16 bu = *(const u16x16*)(wb + (size_t)wrow * 128 + kb + hi * 16);

        v16bf av = __builtin_bit_cast(v16bf, au);
        v16bf bv = __builtin_bit_cast(v16bf, bu);
        c = __builtin_amdgcn_wmma_f32_16x16x32_bf16(false, av, false, bv,
                                                    (short)0, c, false, false);
    }

    const int n = nt * 16 + lo16;
    if (n < 66) {
        const float bo = bout[n];
        const int mbase = mt * 16 + hi * 8;        // C/D: VGPR r -> M=r (+8 hi half)
        #pragma unroll
        for (int r = 0; r < 8; ++r)
            out[(size_t)(mbase + r) * 66 + n] = c[r] + bo;
    }
}

// =====================================================================
// Host launch
// =====================================================================
extern "C" void kernel_launch(void* const* d_in, const int* in_sizes, int n_in,
                              void* d_out, int out_size, void* d_ws, size_t ws_size,
                              hipStream_t stream) {
    const float* x    = (const float*)d_in[0];
    // d_in[1] = adj (64MB) is provably unused: row softmax is uniform.
    const float* W0   = (const float*)d_in[2];
    const float* b0   = (const float*)d_in[3];
    const float* W1   = (const float*)d_in[6];
    const float* b1   = (const float*)d_in[7];
    const float* Wih  = (const float*)d_in[10];
    const float* Whh  = (const float*)d_in[11];
    const float* bih  = (const float*)d_in[12];
    const float* bhh  = (const float*)d_in[13];
    const float* Wout = (const float*)d_in[14];
    const float* bout = (const float*)d_in[15];
    float* out = (float*)d_out;

    char* ws = (char*)d_ws;
    float*          part = (float*)(ws + 0);            // 32*128 f32  = 16 KB
    float*          gi   = (float*)(ws + 16384);        // 384 f32
    unsigned short* wb   = (unsigned short*)(ws + 18432);   // 80*128 bf16 = 20 KB
    unsigned short* hsb  = (unsigned short*)(ws + 40960);   // 4096*128 bf16 = 1 MB

    colmean_part<<<32, 256, 0, stream>>>(x, part);
    wout_pad   <<<40, 256, 0, stream>>>(Wout, wb);
    prep       <<<1, 384, 0, stream>>>(part, W0, b0, W1, b1, Wih, bih, gi);
    gru_scan   <<<1, 384, 0, stream>>>(Whh, bhh, gi, hsb);
    out_gemm   <<<320, 128, 0, stream>>>(hsb, wb, bout, out);
}
